// CompressedLinear_89000312307847
// MI455X (gfx1250) — matrix operand order repaired and checked by reading, not compiled, not run on hardware
//
#include <hip/hip_runtime.h>
#include <hip/hip_bf16.h>

// Problem constants (match reference)
constexpr int TOK   = 32;
constexpr int INK   = 8192;          // K
constexpr int OUTN  = 8192;          // N
constexpr int TILES = OUTN / 16;     // 512 N-tiles
constexpr int KP    = 4;             // K partitions (parallelism for BW saturation)
constexpr int KSEG  = INK / KP;      // 2048
constexpr int WPB   = 8;             // waves per block

typedef _Float16 v16h __attribute__((ext_vector_type(16)));
typedef _Float16 v8h  __attribute__((ext_vector_type(8)));
typedef _Float16 v4h  __attribute__((ext_vector_type(4)));
typedef _Float16 v2h  __attribute__((ext_vector_type(2)));
typedef float    v8f  __attribute__((ext_vector_type(8)));

union AFrag { v16h v; struct { v8h lo; v8h hi; } s; };

// int8 -> f16 (exact), 4 bytes -> 4 halves, using v_perm_b32 magic numbers:
//   0x6400 | (q ^ 0x80)  encodes f16(1024 + q + 128); subtract 1152 -> q.
// Per dword: 1 v_xor + 2 v_perm + 2 v_pk_add (vs ~12 ops via cvt chains).
__device__ __forceinline__ void cvt4_i8_f16(unsigned d, _Float16* dst) {
    d ^= 0x80808080u;
    unsigned lo = __builtin_amdgcn_perm(0x64646464u, d, 0x04010400u); // [b0,64,b1,64]
    unsigned hi = __builtin_amdgcn_perm(0x64646464u, d, 0x04030402u); // [b2,64,b3,64]
    v2h hl = __builtin_bit_cast(v2h, lo);
    v2h hh = __builtin_bit_cast(v2h, hi);
    const v2h bias = {(_Float16)1152.0f, (_Float16)1152.0f};
    hl = hl - bias;
    hh = hh - bias;
    dst[0] = hl[0]; dst[1] = hl[1]; dst[2] = hh[0]; dst[3] = hh[1];
}

// ---------------------------------------------------------------------------
// Kernel 1: input f32 -> f16 into ws, plus per-token sums S[t] (f32, exact
// sums of the f16-rounded values). One block per token, no atomics.
// ---------------------------------------------------------------------------
__global__ __launch_bounds__(256) void prep_kernel(
    const float* __restrict__ in, _Float16* __restrict__ outh,
    float* __restrict__ S)
{
    const int t = blockIdx.x;
    const float4* row  = (const float4*)(in + (size_t)t * INK);
    v4h*          orow = (v4h*)(outh + (size_t)t * INK);

    float s = 0.0f;
    for (int i = threadIdx.x; i < INK / 4; i += blockDim.x) {
        float4 v = row[i];
        v4h h;
        h[0] = (_Float16)v.x; h[1] = (_Float16)v.y;
        h[2] = (_Float16)v.z; h[3] = (_Float16)v.w;
        orow[i] = h;
        s += (float)h[0] + (float)h[1] + (float)h[2] + (float)h[3];
    }

    __shared__ float red[256];
    red[threadIdx.x] = s;
    __syncthreads();
    for (int off = 128; off > 0; off >>= 1) {
        if ((int)threadIdx.x < off) red[threadIdx.x] += red[threadIdx.x + off];
        __syncthreads();
    }
    if (threadIdx.x == 0) S[t] = red[0];
}

// ---------------------------------------------------------------------------
// Kernel 2: K-partitioned skinny GEMM. Wave (kp, tile) accumulates a 2048-deep
// K slice for one 16-wide N tile x both 16-row M tiles; raw C fragments go to
// ws. 2048 waves total -> enough concurrency to saturate 23.3 TB/s.
// ---------------------------------------------------------------------------
__global__ __launch_bounds__(256) void gemm_partial_kernel(
    const _Float16*    __restrict__ Ah,    // [TOK, INK] f16
    const signed char* __restrict__ W,     // [OUTN, INK] int8
    float*             __restrict__ part)  // [KP][TILES][512] f32 raw fragments
{
    const int lane = threadIdx.x & 31;
    const int wave = threadIdx.x >> 5;
    const int gw   = blockIdx.x * WPB + wave;  // 0 .. KP*TILES-1
    const int kp   = gw / TILES;
    const int tile = gw % TILES;
    const int obase = tile * 16;
    const int lo   = lane & 15;
    const int hi   = lane >> 4;
    const int o    = obase + lo;

    const signed char* wptr  = W  + (size_t)o * INK + (size_t)kp * KSEG + hi * 8;
    const _Float16*    a0ptr = Ah + (size_t)lo        * INK + (size_t)kp * KSEG + hi * 8;
    const _Float16*    a1ptr = Ah + (size_t)(16 + lo) * INK + (size_t)kp * KSEG + hi * 8;

    v8f c0 = {};
    v8f c1 = {};

    for (int ko = 0; ko < KSEG; ko += 128) {            // 16 outer iterations
        __builtin_prefetch(wptr + ko + 1024, 0, 1);     // global_prefetch_b8
#pragma unroll
        for (int ku = 0; ku < 128; ku += 32) {
            const int k = ko + ku;

            // --- weight fragment: 16 int8 -> 16 f16 via perm trick ---
            uint2 q0 = *(const uint2*)(wptr + k);
            uint2 q1 = *(const uint2*)(wptr + k + 16);
            v16h b;
            cvt4_i8_f16(q0.x, (_Float16*)&b + 0);
            cvt4_i8_f16(q0.y, (_Float16*)&b + 4);
            cvt4_i8_f16(q1.x, (_Float16*)&b + 8);
            cvt4_i8_f16(q1.y, (_Float16*)&b + 12);

            // --- A fragments (two 16B loads each) ---
            AFrag a0, a1;
            a0.s.lo = *(const v8h*)(a0ptr + k);
            a0.s.hi = *(const v8h*)(a0ptr + k + 16);
            a1.s.lo = *(const v8h*)(a1ptr + k);
            a1.s.hi = *(const v8h*)(a1ptr + k + 16);

            c0 = __builtin_amdgcn_wmma_f32_16x16x32_f16(
                     false, a0.v, false, b, (short)0, c0, false, false);
            c1 = __builtin_amdgcn_wmma_f32_16x16x32_f16(
                     false, a1.v, false, b, (short)0, c1, false, false);
        }
    }

    // Store raw fragments: 64B per lane, fully coalesced.
    float* p = part + ((size_t)(kp * TILES + tile) * 512) + lane * 16;
    *(v8f*)(p)     = c0;
    *(v8f*)(p + 8) = c1;
}

// ---------------------------------------------------------------------------
// Kernel 3: sum the KP partials (fixed order -> deterministic) and apply the
// dequant epilogue: out[m,o] = scale[o]*(acc - zp[o]*S[m]) + bias[o].
// C layout: lanes 0-15 -> N=lane, M=j ; lanes 16-31 -> N=lane-16, M=8+j.
// ---------------------------------------------------------------------------
__global__ __launch_bounds__(256) void reduce_kernel(
    const float*       __restrict__ part,
    const signed char* __restrict__ zp,
    const float*       __restrict__ scale,
    const float*       __restrict__ bias,
    const float*       __restrict__ S,
    float*             __restrict__ out)
{
    const int lane = threadIdx.x & 31;
    const int wave = threadIdx.x >> 5;
    const int tile = blockIdx.x * WPB + wave;
    const int lo   = lane & 15;
    const int hi   = lane >> 4;
    const int o    = tile * 16 + lo;

    v8f c0 = {};
    v8f c1 = {};
#pragma unroll
    for (int kp = 0; kp < KP; ++kp) {
        const float* p = part + ((size_t)(kp * TILES + tile) * 512) + lane * 16;
        c0 += *(const v8f*)(p);
        c1 += *(const v8f*)(p + 8);
    }

    const float zpf = (float)(int)zp[o];
    const float sc  = scale[o];
    const float bo  = bias[o];
    const int mbase = hi * 8;

#pragma unroll
    for (int j = 0; j < 8; ++j) {
        const int m = mbase + j;
        out[(size_t)m        * OUTN + o] = sc * (c0[j] - zpf * S[m])      + bo;
        out[(size_t)(m + 16) * OUTN + o] = sc * (c1[j] - zpf * S[m + 16]) + bo;
    }
}

// ---------------------------------------------------------------------------
extern "C" void kernel_launch(void* const* d_in, const int* in_sizes, int n_in,
                              void* d_out, int out_size, void* d_ws, size_t ws_size,
                              hipStream_t stream)
{
    (void)in_sizes; (void)n_in; (void)out_size; (void)ws_size;

    const float*       inp   = (const float*)d_in[0];        // [32, 8192] f32
    const signed char* W     = (const signed char*)d_in[1];  // [8192, 8192] int8
    const signed char* zp    = (const signed char*)d_in[2];  // [8192] int8
    const float*       scale = (const float*)d_in[3];        // [8192] f32
    const float*       bias  = (const float*)d_in[4];        // [8192] f32
    float*             out   = (float*)d_out;                // [32, 8192] f32

    // ws layout: [0,512K) f16 input | [512K,+128) S | [512K+4K, +4M) partials
    char* ws = (char*)d_ws;
    _Float16* Ah   = (_Float16*)ws;
    float*    S    = (float*)(ws + (size_t)TOK * INK * sizeof(_Float16));
    float*    part = (float*)(ws + (size_t)TOK * INK * sizeof(_Float16) + 4096);

    prep_kernel<<<TOK, 256, 0, stream>>>(inp, Ah, S);

    // KP*TILES = 2048 waves / 8 per block = 256 blocks
    gemm_partial_kernel<<<(KP * TILES) / WPB, 256, 0, stream>>>(Ah, W, part);

    // 512 waves / 8 per block = 64 blocks
    reduce_kernel<<<TILES / WPB, 256, 0, stream>>>(part, zp, scale, bias, S, out);
}
